// HtrgGraphAttentionLayer_32933809226427
// MI455X (gfx1250) — matrix operand order repaired and checked by reading, not compiled, not run on hardware
//
#include <hip/hip_runtime.h>
#include <hip/hip_bf16.h>

#define B_ 128
#define N_ 4096
#define D_ 128
#define NTILES (N_ / 128)

typedef __attribute__((ext_vector_type(16))) _Float16 v16h;
typedef __attribute__((ext_vector_type(8)))  float    v8f;

// ---------- WMMA fragment helpers (V_WMMA_F32_16X16X32_F16) ----------
// A fragment: 16x32 (MxK) f16. Lane L<16 -> M=L, holds K = ks*32 + {0..7, 16..23};
// lane L>=16 -> M=L-16, holds K = ks*32 + {8..15, 24..31}.
__device__ __forceinline__ v16h wmma_load_a_f32(const float* __restrict__ row,
                                                int kstep, int lane) {
  const int k0 = kstep * 32 + ((lane & 16) ? 8 : 0);
  v16h a;
#pragma unroll
  for (int i = 0; i < 8; ++i) {
    a[i]     = (_Float16)row[k0 + i];
    a[i + 8] = (_Float16)row[k0 + 16 + i];
  }
  return a;
}

__device__ __forceinline__ v16h wmma_load_a_f16(const _Float16* __restrict__ row,
                                                int kstep, int lane) {
  const int k0 = kstep * 32 + ((lane & 16) ? 8 : 0);
  v16h a;
#pragma unroll
  for (int i = 0; i < 8; ++i) {
    a[i]     = row[k0 + i];
    a[i + 8] = row[k0 + 16 + i];
  }
  return a;
}

// B fragment: 32x16 (KxN) f16, B[k,j] = W[j,k] (we multiply by W^T).
// Wl = W row-major in f16 (row j = output column j). Lane L<16 -> N=L,
// K = ks*32 + {0..7, 16..23} ; lane L>=16 -> N=L-16, K = ks*32 + {8..15, 24..31}.
__device__ __forceinline__ v16h wmma_load_b_f16(const _Float16* __restrict__ Wl,
                                                int jt, int kstep, int lane) {
  const int col = jt * 16 + (lane & 15);
  const int k0  = kstep * 32 + ((lane & 16) ? 8 : 0);
  const _Float16* p = Wl + col * D_ + k0;
  v16h b;
#pragma unroll
  for (int i = 0; i < 8; ++i) { b[i] = p[i]; b[i + 8] = p[16 + i]; }
  return b;
}

// Branchless tanh: 1 - 2/(e^{2x}+1); saturates to +/-1, no EXEC divergence.
__device__ __forceinline__ float tanh_fast(float x) {
  float e = __expf(2.0f * x);
  return 1.0f - 2.0f * __builtin_amdgcn_rcpf(e + 1.0f);
}

__device__ __forceinline__ float selu_f(float x) {
  const float scale = 1.0507009873554805f;
  const float alpha = 1.6732632423543772f;
  return scale * (x > 0.0f ? x : alpha * (__expf(x) - 1.0f));
}

// ---------------------------------------------------------------------
// Fused pool kernel (one pass over x, online softmax):
//   y[b,:] = softmax_n( tanh(x[b]@Wa^T + ba) @ wa ) @ x[b]
// grid = (B, 2 branches), block = 256 (8 waves). Each WG streams 32 tiles
// of 128 rows: WMMA scores -> running-max rescale -> weighted accumulate
// (tile re-read hits WGP$/L2, so HBM sees x exactly once).
// ---------------------------------------------------------------------
__global__ __launch_bounds__(256) void htrg_pool_fused(
    const float* __restrict__ xT, const float* __restrict__ WaT,
    const float* __restrict__ baT, const float* __restrict__ waT,
    float* __restrict__ yT,
    const float* __restrict__ xS, const float* __restrict__ WaS,
    const float* __restrict__ baS, const float* __restrict__ waS,
    float* __restrict__ yS) {
  __shared__ _Float16 Wlds[D_ * D_];  // Wa rows (= output cols) in f16
  __shared__ float ba_s[D_];
  __shared__ float wa_s[D_];
  __shared__ float s_lds[128];        // tile scores
  __shared__ float w_lds[128];        // tile exp-weights
  __shared__ float redw[8];           // per-wave max
  __shared__ float red[256];          // final reductions

  const int branch = blockIdx.y;
  const float* x   = branch ? xS : xT;
  const float* Wa  = branch ? WaS : WaT;
  const float* ba  = branch ? baS : baT;
  const float* wa  = branch ? waS : waT;
  float*       y   = branch ? yS : yT;

  const int tid = threadIdx.x;
  for (int i = tid; i < D_ * D_; i += 256) Wlds[i] = (_Float16)Wa[i];
  if (tid < D_) { ba_s[tid] = ba[tid]; wa_s[tid] = wa[tid]; }
  __syncthreads();

  const int b    = blockIdx.x;
  const int wave = tid >> 5;
  const int lane = tid & 31;
  const int col  = tid & 127;   // output column for accumulate phase
  const int half = tid >> 7;    // row-half for accumulate phase
  const float* xb = x + (size_t)b * N_ * D_;

  float M = -3.0e38f;   // running max
  float Sacc = 0.0f;    // running exp-sum partial (threads 0..127)
  float Yacc = 0.0f;    // running weighted-sum partial for (col, half)

  for (int t = 0; t < NTILES; ++t) {
    const int n0 = t * 128;

    // ---- score phase: 16-row strip per wave, WMMA over K=128, 8 col tiles
    const float* xrow = xb + (size_t)(n0 + wave * 16 + (lane & 15)) * D_;
    v16h afrag[4];
#pragma unroll
    for (int ks = 0; ks < 4; ++ks) afrag[ks] = wmma_load_a_f32(xrow, ks, lane);

    float sacc[8];
#pragma unroll
    for (int v = 0; v < 8; ++v) sacc[v] = 0.0f;

#pragma unroll
    for (int jt = 0; jt < 8; ++jt) {
      v8f acc = {};
#pragma unroll
      for (int ks = 0; ks < 4; ++ks) {
        v16h bf = wmma_load_b_f16(Wlds, jt, ks, lane);
        acc = __builtin_amdgcn_wmma_f32_16x16x32_f16(
            false, afrag[ks], false, bf, (short)0, acc, false, false);
      }
      const int   c2  = jt * 16 + (lane & 15);
      const float bav = ba_s[c2];
      const float wav = wa_s[c2];
#pragma unroll
      for (int v = 0; v < 8; ++v)
        sacc[v] += tanh_fast(acc[v] + bav) * wav;
    }
    // column reduction within each 16-lane half (masks 1..8 stay in-half)
#pragma unroll
    for (int v = 0; v < 8; ++v) {
#pragma unroll
      for (int m = 1; m <= 8; m <<= 1) sacc[v] += __shfl_xor(sacc[v], m, 32);
    }
    if ((lane & 15) == 0) {
      const int rbase = wave * 16 + ((lane >> 4) ? 8 : 0);
#pragma unroll
      for (int v = 0; v < 8; ++v) s_lds[rbase + v] = sacc[v];
    }

    // prefetch next tile (gfx1250 global_prefetch_b8); 2 cachelines/thread
    if (t + 1 < NTILES) {
#pragma unroll
      for (int q = 0; q < 2; ++q) {
        const int l = tid * 2 + q;  // 512 lines of 128B cover the 64KB tile
        __builtin_prefetch(xb + (size_t)(n0 + 128 + (l >> 2)) * D_ + (l & 3) * 32,
                           0, 1);
      }
    }
    __syncthreads();  // (1) s_lds ready; prior w_lds readers done

    // ---- tile max (wave shfl + 8-entry LDS combine)
    float mt = (tid < 128) ? s_lds[tid] : -3.0e38f;
#pragma unroll
    for (int m2 = 1; m2 < 32; m2 <<= 1) mt = fmaxf(mt, __shfl_xor(mt, m2, 32));
    if (lane == 0) redw[wave] = mt;
    __syncthreads();  // (2)
    mt = redw[0];
#pragma unroll
    for (int q = 1; q < 8; ++q) mt = fmaxf(mt, redw[q]);

    // ---- online softmax rescale
    const float Mn = fmaxf(M, mt);
    const float r  = __expf(M - Mn);
    M = Mn;
    Yacc *= r;
    if (tid < 128) {
      const float w = __expf(s_lds[tid] - M);
      w_lds[tid] = w;
      Sacc = Sacc * r + w;
    }
    __syncthreads();  // (3) w_lds ready

    // ---- weighted accumulate: rows [half*64, half*64+64), column `col`
    const float* xc = xb + (size_t)(n0 + half * 64) * D_ + col;
#pragma unroll 4
    for (int n = 0; n < 64; ++n)
      Yacc += w_lds[half * 64 + n] * xc[(size_t)n * D_];
  }

  // ---- epilogue: total exp-sum, combine halves, normalize
  __syncthreads();
  red[tid] = (tid < 128) ? Sacc : 0.0f;
  __syncthreads();
  for (int s = 128; s > 0; s >>= 1) {
    if (tid < s) red[tid] += red[tid + s];
    __syncthreads();
  }
  const float inv = 1.0f / red[0];
  __syncthreads();
  red[tid] = Yacc;
  __syncthreads();
  if (tid < 128)
    y[(size_t)b * D_ + col] = (red[tid] + red[tid + 128]) * inv;
}

// ---------------------------------------------------------------------
// Final kernel: Z = (yT + yS + master) @ Wp^T ; pre = Z + 3*bp ;
//   out_{T,S,M} = selu(pre * g/sqrt(1+eps) + beta). One WG, 8 waves.
// ---------------------------------------------------------------------
__global__ __launch_bounds__(256) void htrg_final_kernel(
    const float* __restrict__ yT, const float* __restrict__ yS,
    const float* __restrict__ master, const float* __restrict__ Wp,
    const float* __restrict__ bp, const float* __restrict__ gT,
    const float* __restrict__ betaT, const float* __restrict__ gS,
    const float* __restrict__ betaS, const float* __restrict__ gM,
    const float* __restrict__ betaM, float* __restrict__ out) {
  __shared__ _Float16 Wlds[D_ * D_];
  __shared__ _Float16 Ulds[B_ * D_];

  const int tid = threadIdx.x;
  for (int i = tid; i < D_ * D_; i += 256) {
    Wlds[i] = (_Float16)Wp[i];
    Ulds[i] = (_Float16)(yT[i] + yS[i] + master[i]);
  }
  __syncthreads();

  const int wave = tid >> 5;
  const int lane = tid & 31;
  const int m0   = wave * 16;  // batch rows for this wave
  const _Float16* urow = Ulds + (m0 + (lane & 15)) * D_;
  const float invs = rsqrtf(1.0f + 1e-5f);  // BN eval scale

  v16h afrag[4];
#pragma unroll
  for (int ks = 0; ks < 4; ++ks) afrag[ks] = wmma_load_a_f16(urow, ks, lane);

#pragma unroll
  for (int jt = 0; jt < 8; ++jt) {
    v8f acc = {};
#pragma unroll
    for (int ks = 0; ks < 4; ++ks) {
      v16h bf = wmma_load_b_f16(Wlds, jt, ks, lane);
      acc = __builtin_amdgcn_wmma_f32_16x16x32_f16(
          false, afrag[ks], false, bf, (short)0, acc, false, false);
    }
    const int   col = jt * 16 + (lane & 15);
    const float bpv = 3.0f * bp[col];
    const float sT  = gT[col] * invs, bT = betaT[col];
    const float sS  = gS[col] * invs, bS = betaS[col];
    const float sM  = gM[col] * invs, bM = betaM[col];
#pragma unroll
    for (int v = 0; v < 8; ++v) {
      const int row = m0 + v + ((lane & 16) ? 8 : 0);
      const float pre = acc[v] + bpv;
      out[(size_t)row * D_ + col]                 = selu_f(pre * sT + bT);
      out[(size_t)(B_ * D_) + row * D_ + col]     = selu_f(pre * sS + bS);
      out[(size_t)(2 * B_ * D_) + row * D_ + col] = selu_f(pre * sM + bM);
    }
  }
}

// ---------------------------------------------------------------------
extern "C" void kernel_launch(void* const* d_in, const int* in_sizes, int n_in,
                              void* d_out, int out_size, void* d_ws,
                              size_t ws_size, hipStream_t stream) {
  const float* x_T    = (const float*)d_in[0];
  const float* x_S    = (const float*)d_in[1];
  const float* master = (const float*)d_in[2];
  const float* WaT    = (const float*)d_in[3];
  const float* baT    = (const float*)d_in[4];
  const float* waT    = (const float*)d_in[5];
  const float* WaS    = (const float*)d_in[6];
  const float* baS    = (const float*)d_in[7];
  const float* waS    = (const float*)d_in[8];
  // d_in[9..11] (WaM, baM, waM) are dead: master has N=1 -> softmax == 1.
  const float* Wp     = (const float*)d_in[12];
  const float* bp     = (const float*)d_in[13];
  const float* gT     = (const float*)d_in[14];
  const float* betaT  = (const float*)d_in[15];
  const float* gS     = (const float*)d_in[16];
  const float* betaS  = (const float*)d_in[17];
  const float* gM     = (const float*)d_in[18];
  const float* betaM  = (const float*)d_in[19];

  float* ws = (float*)d_ws;
  float* yT = ws;                    // B*D
  float* yS = yT + (size_t)B_ * D_;  // B*D

  dim3 g(B_, 2);
  htrg_pool_fused<<<g, 256, 0, stream>>>(x_T, WaT, baT, waT, yT,
                                         x_S, WaS, baS, waS, yS);
  htrg_final_kernel<<<1, 256, 0, stream>>>(yT, yS, master, Wp, bp, gT, betaT,
                                           gS, betaS, gM, betaM,
                                           (float*)d_out);
}
